// TemporalHybrid_34402688041060
// MI455X (gfx1250) — matrix operand order repaired and checked by reading, not compiled, not run on hardware
//
#include <hip/hip_runtime.h>
#include <hip/hip_bf16.h>

#define B_    8
#define L_    1024
#define IND_  128
#define H_    256
#define N_    256
#define MTOT_ (B_*L_)   // 8192

typedef __bf16 bf16;
typedef bf16  v16bf __attribute__((ext_vector_type(16)));
typedef bf16  v8bf  __attribute__((ext_vector_type(8)));
typedef float v8f   __attribute__((ext_vector_type(8)));

// ---------------------------------------------------------------- helpers

__device__ __forceinline__ float wave_sum32(float v) {
#pragma unroll
  for (int off = 16; off > 0; off >>= 1) v += __shfl_xor(v, off, 32);
  return v;
}

__device__ __forceinline__ float gelu_tanh(float x) {
  float x3 = x * x * x;
  return 0.5f * x * (1.0f + tanhf(0.7978845608028654f * (x + 0.044715f * x3)));
}

// A-fragment element->K map (contiguous order, matches B fragments):
// lanes 0-15: K = e. lanes 16-31: K = 16 + e.
__device__ __forceinline__ int kmapA(int hi, int e) {
  return hi * 16 + e;
}

__device__ __forceinline__ v16bf concat16(v8bf lo, v8bf hh) {
  return __builtin_shufflevector(lo, hh, 0, 1, 2, 3, 4, 5, 6, 7,
                                         8, 9, 10, 11, 12, 13, 14, 15);
}

// A fragment from a row-major bf16 row (at &row[ktile*32]); two 16B loads.
__device__ __forceinline__ v16bf load_a_row(const bf16* rowk, int hi) {
  v8bf lo = *(const v8bf*)(rowk + hi * 16);
  v8bf hh = *(const v8bf*)(rowk + hi * 16 + 8);
  return concat16(lo, hh);
}

// B fragment when the lane's 16 consecutive K values are contiguous in memory
// (weights pre-transposed to [N][K]); two 16B loads.
__device__ __forceinline__ v16bf load_frag_contig(const bf16* p) {
  v8bf lo = *(const v8bf*)p;
  v8bf hh = *(const v8bf*)(p + 8);
  return concat16(lo, hh);
}

__device__ __forceinline__ v16bf zero_frag() {
  v16bf a;
#pragma unroll
  for (int e = 0; e < 16; e++) a[e] = (bf16)0.0f;
  return a;
}

// ---------------------------------------------------------------- converts

__global__ void k_cvt_bf16(const float* __restrict__ in, bf16* __restrict__ out, int n) {
  int i = blockIdx.x * blockDim.x + threadIdx.x;
  if (i < n) out[i] = (bf16)in[i];
}

// out[c][r] = in[r][c]  (f32 [R,C] -> bf16 [C,R])
__global__ void k_cvt_t_bf16(const float* __restrict__ in, bf16* __restrict__ out,
                             int R, int C) {
  int i = blockIdx.x * blockDim.x + threadIdx.x;
  if (i < R * C) {
    int r = i / C, c = i % C;
    out[(size_t)c * R + r] = (bf16)in[i];
  }
}

// W_tcn [O,I,3] f32 -> [3,O,I] bf16
__global__ void k_cvt_tcn(const float* __restrict__ in, bf16* __restrict__ out) {
  int i = blockIdx.x * blockDim.x + threadIdx.x;
  if (i < H_ * IND_ * 3) {
    int kk = i % 3;
    int oi = i / 3;              // o*IND + i_in
    out[(size_t)kk * H_ * IND_ + oi] = (bf16)in[i];
  }
}

// ---------------------------------------------------------------- kernels

// S4D kernel generation: K[h,l] = 2*Re(sum_n CB[h,n] * r[h,n]^l), via recurrence.
__global__ void k_s4d_kernel(const float* __restrict__ log_dt, const float* __restrict__ Alr,
                             const float* __restrict__ Aim, const float* __restrict__ Bre,
                             const float* __restrict__ Bim, const float* __restrict__ Cre,
                             const float* __restrict__ Cim, float* __restrict__ Kout) {
  __shared__ float sre[8][N_];
  int h = blockIdx.x;
  int n = threadIdx.x;
  int lane = n & 31, wid = n >> 5;
  size_t idx = (size_t)h * N_ + n;
  float dt  = expf(log_dt[h]);
  float are = -expf(Alr[idx]);
  float aim = Aim[idx];
  float dre = dt * are, dimv = dt * aim;
  float er  = expf(dre);
  float rre = er * cosf(dimv), rim = er * sinf(dimv);   // r = exp(dt*A)
  float ere = rre - 1.0f, eim = rim;                    // r - 1
  float a2  = are * are + aim * aim;
  float qre = (ere * are + eim * aim) / a2;             // (r-1)/A
  float qim = (eim * are - ere * aim) / a2;
  float bre = Bre[idx], bim = Bim[idx];
  float bbre = qre * bre - qim * bim;
  float bbim = qre * bim + qim * bre;
  float cre = Cre[idx], cim = Cim[idx];
  float sr = cre * bbre - cim * bbim;                   // s = C*Bbar
  float si = cre * bbim + cim * bbre;
  for (int l0 = 0; l0 < L_; l0 += 8) {
#pragma unroll
    for (int j = 0; j < 8; j++) {
      sre[j][n] = sr;
      float nsr = sr * rre - si * rim;
      float nsi = sr * rim + si * rre;
      sr = nsr; si = nsi;
    }
    __syncthreads();
    float acc = 0.0f;
#pragma unroll
    for (int c = 0; c < 8; c++) acc += sre[wid][lane + c * 32];
    acc = wave_sum32(acc);
    if (lane == 0) Kout[(size_t)h * L_ + l0 + wid] = 2.0f * acc;
    __syncthreads();
  }
}

// Encoder: h0 = x @ W_enc + b_enc. Also emits u_bf16 = h0^T in [B,H,L] (bf16).
// W_enc pre-transposed to [H, IND] bf16.
__global__ void k_encoder(const bf16* __restrict__ xb, const bf16* __restrict__ wencT,
                          const float* __restrict__ bias,
                          float* __restrict__ h0, bf16* __restrict__ u_bf) {
  int tid = threadIdx.x;
  int lane = tid & 31, wid = tid >> 5;
  int waveId = blockIdx.x * 8 + wid;
  int mt = waveId >> 4;              // 0..511
  int nt = waveId & 15;              // 0..15
  int hi = lane >> 4, i16 = lane & 15;
  int mrow = mt * 16 + i16;
  int ncol = nt * 16 + i16;
  v8f c;
#pragma unroll
  for (int v = 0; v < 8; v++) c[v] = 0.0f;
#pragma unroll
  for (int kt = 0; kt < 4; kt++) {
    v16bf a = load_a_row(xb + (size_t)mrow * IND_ + kt * 32, hi);
    v16bf b = load_frag_contig(wencT + (size_t)ncol * IND_ + kt * 32 + hi * 16);
    c = __builtin_amdgcn_wmma_f32_16x16x32_bf16(false, a, false, b, (short)0, c, false, false);
  }
  float bv = bias[ncol];
#pragma unroll
  for (int v = 0; v < 8; v++) {
    int m = mt * 16 + v + hi * 8;
    float val = c[v] + bv;
    h0[(size_t)m * H_ + ncol] = val;
    int bb = m >> 10, l = m & 1023;
    u_bf[((size_t)bb * H_ + ncol) * L_ + l] = (bf16)val;
  }
}

// Causal conv as lower-triangular Toeplitz GEMM: Y[l,b] = sum_j K[l-j] U[j,b].
// K row staged GLOBAL->LDS with the gfx1250 async path, then converted once
// into a zero-padded bf16 table so the Toeplitz A-fragment gather is
// branch-free ds_load_u16.
__global__ void k_s4_conv(const bf16* __restrict__ u_bf, const float* __restrict__ Kf,
                          const float* __restrict__ h0, const float* __restrict__ Dv,
                          bf16* __restrict__ g_bf) {
  __shared__ __align__(16) float Ksf[L_];
  __shared__ __align__(16) bf16  Kpad[64 + L_];
  int tid = threadIdx.x;
  int h = blockIdx.x >> 3;
  int grp = blockIdx.x & 7;

  // 4 KB async copy: 256 lanes x 16 B, memory -> LDS, no VGPR round-trip.
  {
    unsigned ldsoff = (unsigned)(size_t)(&Ksf[0]) + (unsigned)tid * 16u;
    const float* gsrc = Kf + (size_t)h * L_ + tid * 4;
    asm volatile("global_load_async_to_lds_b128 %0, %1, off"
                 :: "v"(ldsoff), "v"(gsrc) : "memory");
    asm volatile("s_wait_asynccnt 0" ::: "memory");
  }
  __syncthreads();
  // one-time bf16 conversion with 64-element zero apron
  for (int i = tid; i < L_; i += 256) Kpad[64 + i] = (bf16)Ksf[i];
  if (tid < 64) Kpad[tid] = (bf16)0.0f;
  __syncthreads();

  int lane = tid & 31, wid = tid >> 5;
  int mt = grp * 8 + wid;            // 0..63
  int hi = lane >> 4, i16 = lane & 15;
  int mrow = mt * 16 + i16;
  int bcol = i16;                    // batch column, valid < 8
  v8f c;
#pragma unroll
  for (int v = 0; v < 8; v++) c[v] = 0.0f;
  int jtmax = (mt * 16 + 15) >> 5;
  const bf16* ubase = u_bf + ((size_t)((bcol < 8) ? bcol : 0) * H_ + h) * L_;
  for (int jt = 0; jt <= jtmax; jt++) {
    int abase = 64 + mrow - jt * 32;         // Kpad index for kk = 0
    v16bf a;
#pragma unroll
    for (int e = 0; e < 16; e++) a[e] = Kpad[abase - kmapA(hi, e)];
    v16bf b;
    if (bcol < 8) {
      b = load_frag_contig(ubase + jt * 32 + hi * 16);
    } else {
      b = zero_frag();
    }
    c = __builtin_amdgcn_wmma_f32_16x16x32_bf16(false, a, false, b, (short)0, c, false, false);
  }
  float dh = Dv[h];
  if (bcol < 8) {
#pragma unroll
    for (int v = 0; v < 8; v++) {
      int l = mt * 16 + v + hi * 8;
      float uf = h0[((size_t)bcol * L_ + l) * H_ + h];
      float y = c[v] + dh * uf;
      g_bf[((size_t)bcol * L_ + l) * H_ + h] = (bf16)gelu_tanh(y);
    }
  }
}

// TCN: Conv1d(in_dim->H, k=3, dilation=2, pad=2) == 3 shifted GEMMs on x_bf16.
// W_tcn pre-laid-out as [3][O][I] bf16.
__global__ void k_tcn(const bf16* __restrict__ xb, const bf16* __restrict__ wtcnT,
                      const float* __restrict__ bias, float* __restrict__ tcn) {
  int tid = threadIdx.x;
  int lane = tid & 31, wid = tid >> 5;
  int waveId = blockIdx.x * 8 + wid;
  int mt = waveId >> 4, nt = waveId & 15;
  int hi = lane >> 4, i16 = lane & 15;
  int m = mt * 16 + i16;
  int bb = m >> 10, l = m & 1023;
  int ncol = nt * 16 + i16;
  v8f c;
#pragma unroll
  for (int v = 0; v < 8; v++) c[v] = 0.0f;
#pragma unroll
  for (int k = 0; k < 3; k++) {
    int srcl = l + 2 * k - 2;
    bool valid = (srcl >= 0) && (srcl < L_);
    const bf16* wk = wtcnT + (size_t)k * H_ * IND_ + (size_t)ncol * IND_;
#pragma unroll
    for (int kt = 0; kt < 4; kt++) {
      v16bf a;
      if (valid) {
        a = load_a_row(xb + ((size_t)bb * L_ + srcl) * IND_ + kt * 32, hi);
      } else {
        a = zero_frag();
      }
      v16bf b = load_frag_contig(wk + kt * 32 + hi * 16);
      c = __builtin_amdgcn_wmma_f32_16x16x32_bf16(false, a, false, b, (short)0, c, false, false);
    }
  }
  float bv = bias[ncol];
#pragma unroll
  for (int v = 0; v < 8; v++)
    tcn[(size_t)(mt * 16 + v + hi * 8) * H_ + ncol] = c[v] + bv;
}

// Fused tail: z = g@Wblk+b_blk (+h0) -> LN1 -> h1 -> @Wdec+b_dec -> +tcn
//             -> nan_to_num -> LN2 -> out.  One block per 16-row tile.
// Wblk/Wdec pre-transposed to [N][K] bf16.
__global__ void k_fuse(const bf16* __restrict__ g_bf, const bf16* __restrict__ wblkT,
                       const float* __restrict__ b_blk, const float* __restrict__ h0,
                       const float* __restrict__ ln1g, const float* __restrict__ ln1b,
                       const bf16* __restrict__ wdecT, const float* __restrict__ b_dec,
                       const float* __restrict__ tcn,
                       const float* __restrict__ ln2g, const float* __restrict__ ln2b,
                       float* __restrict__ out) {
  __shared__ float zt[16][H_];
  __shared__ __align__(16) bf16 h1t[16][H_];
  int tid = threadIdx.x, lane = tid & 31, wid = tid >> 5;
  int hi = lane >> 4, i16 = lane & 15;
  int m0 = blockIdx.x * 16;
  const float inv_h = 1.0f / (float)H_;

  // phase 1: z + h0 -> zt
#pragma unroll
  for (int t = 0; t < 2; t++) {
    int nt = wid * 2 + t;
    int ncol = nt * 16 + i16;
    v8f c;
#pragma unroll
    for (int v = 0; v < 8; v++) c[v] = 0.0f;
#pragma unroll
    for (int kt = 0; kt < 8; kt++) {
      v16bf a = load_a_row(g_bf + (size_t)(m0 + i16) * H_ + kt * 32, hi);
      v16bf b = load_frag_contig(wblkT + (size_t)ncol * H_ + kt * 32 + hi * 16);
      c = __builtin_amdgcn_wmma_f32_16x16x32_bf16(false, a, false, b, (short)0, c, false, false);
    }
    float bv = b_blk[ncol];
#pragma unroll
    for (int v = 0; v < 8; v++) {
      int r = v + hi * 8;
      zt[r][ncol] = c[v] + bv + h0[(size_t)(m0 + r) * H_ + ncol];
    }
  }
  __syncthreads();

  // phase 2: LN1 -> h1t (bf16). wave w handles rows {w, w+8}
#pragma unroll
  for (int rr = 0; rr < 2; rr++) {
    int r = wid + rr * 8;
    float s = 0.0f, s2 = 0.0f;
    for (int cc = lane; cc < H_; cc += 32) { float x = zt[r][cc]; s += x; s2 += x * x; }
    s = wave_sum32(s); s2 = wave_sum32(s2);
    float mean = s * inv_h;
    float var = s2 * inv_h - mean * mean;
    float inv = rsqrtf(var + 1e-5f);
    for (int cc = lane; cc < H_; cc += 32)
      h1t[r][cc] = (bf16)((zt[r][cc] - mean) * inv * ln1g[cc] + ln1b[cc]);
  }
  __syncthreads();

  // phase 3: s4_out = h1 @ Wdec + b_dec; + tcn; nan_to_num -> zt
#pragma unroll
  for (int t = 0; t < 2; t++) {
    int nt = wid * 2 + t;
    int ncol = nt * 16 + i16;
    v8f c;
#pragma unroll
    for (int v = 0; v < 8; v++) c[v] = 0.0f;
#pragma unroll
    for (int kt = 0; kt < 8; kt++) {
      // h1 tile is row-major contiguous in K -> two 16B ds_loads
      const bf16* ap = &h1t[i16][kt * 32];
      v16bf a = load_a_row(ap, hi);
      v16bf b = load_frag_contig(wdecT + (size_t)ncol * H_ + kt * 32 + hi * 16);
      c = __builtin_amdgcn_wmma_f32_16x16x32_bf16(false, a, false, b, (short)0, c, false, false);
    }
    float bv = b_dec[ncol];
#pragma unroll
    for (int v = 0; v < 8; v++) {
      int r = v + hi * 8;
      float x = c[v] + bv + tcn[(size_t)(m0 + r) * H_ + ncol];
      if (__builtin_isnan(x)) x = 0.0f;
      else if (__builtin_isinf(x)) x = (x > 0.0f) ? 1e6f : -1e6f;
      zt[r][ncol] = x;
    }
  }
  __syncthreads();

  // phase 4: LN2 -> out
#pragma unroll
  for (int rr = 0; rr < 2; rr++) {
    int r = wid + rr * 8;
    float s = 0.0f, s2 = 0.0f;
    for (int cc = lane; cc < H_; cc += 32) { float x = zt[r][cc]; s += x; s2 += x * x; }
    s = wave_sum32(s); s2 = wave_sum32(s2);
    float mean = s * inv_h;
    float var = s2 * inv_h - mean * mean;
    float inv = rsqrtf(var + 1e-5f);
    for (int cc = lane; cc < H_; cc += 32)
      out[(size_t)(m0 + r) * H_ + cc] = (zt[r][cc] - mean) * inv * ln2g[cc] + ln2b[cc];
  }
}

// ---------------------------------------------------------------- launch

extern "C" void kernel_launch(void* const* d_in, const int* in_sizes, int n_in,
                              void* d_out, int out_size, void* d_ws, size_t ws_size,
                              hipStream_t stream) {
  (void)in_sizes; (void)n_in; (void)out_size; (void)ws_size;
  const float* x      = (const float*)d_in[0];
  const float* W_enc  = (const float*)d_in[1];
  const float* b_enc  = (const float*)d_in[2];
  const float* log_dt = (const float*)d_in[3];
  const float* Alr    = (const float*)d_in[4];
  const float* Aim    = (const float*)d_in[5];
  const float* Bre    = (const float*)d_in[6];
  const float* Bim    = (const float*)d_in[7];
  const float* Cre    = (const float*)d_in[8];
  const float* Cim    = (const float*)d_in[9];
  const float* Dv     = (const float*)d_in[10];
  const float* W_blk  = (const float*)d_in[11];
  const float* b_blk  = (const float*)d_in[12];
  const float* ln1g   = (const float*)d_in[13];
  const float* ln1b   = (const float*)d_in[14];
  const float* W_dec  = (const float*)d_in[15];
  const float* b_dec  = (const float*)d_in[16];
  const float* W_tcn  = (const float*)d_in[17];
  const float* b_tcn  = (const float*)d_in[18];
  const float* ln2g   = (const float*)d_in[19];
  const float* ln2b   = (const float*)d_in[20];

  char* w = (char*)d_ws;
  bf16*  xb     = (bf16*) (w + 0);          //  2,097,152 B  x bf16 [8192,128]
  bf16*  wencT  = (bf16*) (w + 2097152);    //     65,536 B  W_enc^T bf16 [256,128]
  bf16*  wblkT  = (bf16*) (w + 2162688);    //    131,072 B  W_blk^T bf16 [256,256]
  bf16*  wdecT  = (bf16*) (w + 2293760);    //    131,072 B  W_dec^T bf16 [256,256]
  bf16*  wtcnT  = (bf16*) (w + 2424832);    //    196,608 B  W_tcn bf16 [3,256,128]
  float* Kbuf   = (float*)(w + 2621440);    //  1,048,576 B  K [256,1024] f32
  float* h0     = (float*)(w + 3670016);    //  8,388,608 B  h0 [8192,256] f32
  bf16*  ubf    = (bf16*) (w + 12058624);   //  4,194,304 B  u  [8,256,1024] bf16
  bf16*  gbf    = (bf16*) (w + 16252928);   //  4,194,304 B  gelu(y) [8192,256] bf16
  float* tcnbuf = (float*)(w + 20447232);   //  8,388,608 B  tcn [8192,256] f32
  // total 28,835,840 bytes

  k_cvt_bf16 <<<(MTOT_*IND_ + 255) / 256, 256, 0, stream>>>(x, xb, MTOT_*IND_);
  k_cvt_t_bf16<<<(IND_*H_  + 255) / 256, 256, 0, stream>>>(W_enc, wencT, IND_, H_);
  k_cvt_t_bf16<<<(H_*H_    + 255) / 256, 256, 0, stream>>>(W_blk, wblkT, H_, H_);
  k_cvt_t_bf16<<<(H_*H_    + 255) / 256, 256, 0, stream>>>(W_dec, wdecT, H_, H_);
  k_cvt_tcn  <<<(H_*IND_*3 + 255) / 256, 256, 0, stream>>>(W_tcn, wtcnT);

  k_s4d_kernel<<<H_, N_, 0, stream>>>(log_dt, Alr, Aim, Bre, Bim, Cre, Cim, Kbuf);
  k_encoder   <<<1024, 256, 0, stream>>>(xb, wencT, b_enc, h0, ubf);
  k_s4_conv   <<<2048, 256, 0, stream>>>(ubf, Kbuf, h0, Dv, gbf);
  k_tcn       <<<1024, 256, 0, stream>>>(xb, wtcnT, b_tcn, tcnbuf);
  k_fuse      <<<512, 256, 0, stream>>>(gbf, wblkT, b_blk, h0, ln1g, ln1b,
                                        wdecT, b_dec, tcnbuf, ln2g, ln2b, (float*)d_out);
}
